// GNN_31061203485251
// MI455X (gfx1250) — matrix operand order repaired and checked by reading, not compiled, not run on hardware
//
#include <hip/hip_runtime.h>
#include <hip/hip_bf16.h>
#include <math.h>

// ---------------- problem constants (fixed by reference) ----------------
#define NN  100000     // nodes
#define NE  1600000    // edges
#define NB  1000       // graphs
#define DIN 4
#define DD  128
#define NPG (NN / NB)  // 100 nodes per graph, contiguous (batch = i/100)
#define GEPS 1e-5f
#define NBLK ((NN + 255) / 256)   // 391 scan blocks

typedef _Float16 v16h __attribute__((ext_vector_type(16)));
typedef _Float16 v8h  __attribute__((ext_vector_type(8)));
typedef float    v8f  __attribute__((ext_vector_type(8)));

__device__ __forceinline__ float wave_red_sum(float v) {
  #pragma unroll
  for (int m = 16; m > 0; m >>= 1) v += __shfl_xor(v, m);
  return v;
}
__device__ __forceinline__ float sigm(float x) { return 1.0f / (1.0f + expf(-x)); }

// ---------------- generic zero fill (works for int buffers too: 0.0f == 0x0) ----
__global__ void k_zero(float* __restrict__ p, long long n) {
  long long i = (long long)blockIdx.x * blockDim.x + threadIdx.x;
  if (i < n) p[i] = 0.0f;
}

// ---------------- GraphNorm0 (whole-batch, D_IN=4) ----------------
__global__ void k_gn0_sum(const float* __restrict__ x, float* __restrict__ stats) {
  int n = blockIdx.x * blockDim.x + threadIdx.x;
  float4 v = make_float4(0.f, 0.f, 0.f, 0.f);
  if (n < NN) v = *(const float4*)(x + (size_t)n * 4);
  float s0 = wave_red_sum(v.x), s1 = wave_red_sum(v.y);
  float s2 = wave_red_sum(v.z), s3 = wave_red_sum(v.w);
  if ((threadIdx.x & 31) == 0) {
    atomicAdd(stats + 0, s0); atomicAdd(stats + 1, s1);
    atomicAdd(stats + 2, s2); atomicAdd(stats + 3, s3);
  }
}
__global__ void k_gn0_var(const float* __restrict__ x, const float* __restrict__ a,
                          float* __restrict__ stats) {
  int n = blockIdx.x * blockDim.x + threadIdx.x;
  float t[4] = {0.f, 0.f, 0.f, 0.f};
  if (n < NN) {
    float4 v = *(const float4*)(x + (size_t)n * 4);
    const float inv = 1.0f / (float)NN;
    t[0] = v.x - a[0] * stats[0] * inv; t[1] = v.y - a[1] * stats[1] * inv;
    t[2] = v.z - a[2] * stats[2] * inv; t[3] = v.w - a[3] * stats[3] * inv;
  }
  #pragma unroll
  for (int k = 0; k < 4; ++k) {
    float s = wave_red_sum(t[k] * t[k]);
    if ((threadIdx.x & 31) == 0) atomicAdd(stats + 4 + k, s);
  }
}
__global__ void k_gn0_norm(const float* __restrict__ x, const float* __restrict__ w,
                           const float* __restrict__ b, const float* __restrict__ a,
                           const float* __restrict__ stats, float* __restrict__ out) {
  int n = blockIdx.x * blockDim.x + threadIdx.x;
  if (n >= NN) return;
  const float inv = 1.0f / (float)NN;
  float4 v = *(const float4*)(x + (size_t)n * 4);
  float o[4]; float vv[4] = {v.x, v.y, v.z, v.w};
  #pragma unroll
  for (int k = 0; k < 4; ++k) {
    float t = vv[k] - a[k] * stats[k] * inv;
    o[k] = w[k] * t * rsqrtf(stats[4 + k] * inv + GEPS) + b[k];
  }
  *(float4*)(out + (size_t)n * 4) = make_float4(o[0], o[1], o[2], o[3]);
}

// ---------------- conv1 dense: h = xn0 @ W1  (K=4, VALU) ----------------
__global__ __launch_bounds__(DD) void k_conv1(const float* __restrict__ xn0,
                                              const float* __restrict__ W1,
                                              float* __restrict__ h) {
  int n = blockIdx.x, d = threadIdx.x;
  float4 xv = *(const float4*)(xn0 + (size_t)n * 4);
  float s = xv.x * W1[d] + xv.y * W1[DD + d] + xv.z * W1[2 * DD + d] + xv.w * W1[3 * DD + d];
  h[(size_t)n * DD + d] = s;
}

// ---------------- CSR build: degree, scan, dinv, fill ----------------
__global__ void k_deg_i(const int* __restrict__ dst, int* __restrict__ degi) {
  int e = blockIdx.x * blockDim.x + threadIdx.x;
  if (e < NE) atomicAdd(degi + dst[e], 1);
}
__global__ void k_dinv_i(const int* __restrict__ degi, float* __restrict__ dinv) {
  int i = blockIdx.x * blockDim.x + threadIdx.x;
  if (i < NN) dinv[i] = rsqrtf((float)degi[i] + 1.0f);
}
// block-local exclusive scan (Hillis-Steele in LDS) + block totals
__global__ __launch_bounds__(256) void k_scan1(const int* __restrict__ degi,
                                               int* __restrict__ rows,
                                               int* __restrict__ bsum) {
  __shared__ int sh[256];
  int t = threadIdx.x;
  int i = blockIdx.x * 256 + t;
  int v = (i < NN) ? degi[i] : 0;
  sh[t] = v;
  __syncthreads();
  #pragma unroll
  for (int off = 1; off < 256; off <<= 1) {
    int x = (t >= off) ? sh[t - off] : 0;
    __syncthreads();
    sh[t] += x;
    __syncthreads();
  }
  if (i < NN) rows[i] = sh[t] - v;           // exclusive within block
  if (t == 255) bsum[blockIdx.x] = sh[255];
}
__global__ __launch_bounds__(512) void k_scan2(const int* __restrict__ bsum,
                                               int* __restrict__ boff, int nb) {
  __shared__ int sh[512];
  int t = threadIdx.x;
  int v = (t < nb) ? bsum[t] : 0;
  sh[t] = v;
  __syncthreads();
  #pragma unroll
  for (int off = 1; off < 512; off <<= 1) {
    int x = (t >= off) ? sh[t - off] : 0;
    __syncthreads();
    sh[t] += x;
    __syncthreads();
  }
  if (t < nb) boff[t] = sh[t] - v;           // exclusive
}
__global__ void k_scan3(int* __restrict__ rows, const int* __restrict__ boff) {
  int i = blockIdx.x * blockDim.x + threadIdx.x;
  if (i < NN) rows[i] += boff[i >> 8];
  if (i == 0) rows[NN] = NE;
}
__global__ void k_fill(const int* __restrict__ src, const int* __restrict__ dst,
                       const int* __restrict__ rows, int* __restrict__ cursor,
                       const float* __restrict__ dinv,
                       int* __restrict__ csrc, float* __restrict__ cw) {
  int e = blockIdx.x * blockDim.x + threadIdx.x;
  if (e >= NE) return;
  int s = src[e], t = dst[e];
  int slot = atomicAdd(cursor + t, 1);
  int pos = rows[t] + slot;
  csrc[pos] = s;
  cw[pos] = dinv[s] * dinv[t];
}

// ---------------- CSR gather: out[n] = sum_e w*h[src] + h[n]*dinv^2 + bias ----------------
// one wave per destination node; float4 per lane (128 dims / 32 lanes)
__global__ __launch_bounds__(256) void k_gather(const float* __restrict__ h,
                                                const int* __restrict__ rows,
                                                const int* __restrict__ csrc,
                                                const float* __restrict__ cw,
                                                const float* __restrict__ dinv,
                                                const float* __restrict__ bias,
                                                float* __restrict__ out) {
  long long gid = (long long)blockIdx.x * blockDim.x + threadIdx.x;
  int n = (int)(gid >> 5);
  if (n >= NN) return;
  int lane = (int)(gid & 31);
  int beg = rows[n], end = rows[n + 1];
  float a0 = 0.f, a1 = 0.f, a2 = 0.f, a3 = 0.f;
  for (int p = beg; p < end; ++p) {
    int s = csrc[p];
    float w = cw[p];
    float4 hv = *(const float4*)(h + (size_t)s * DD + lane * 4);
    a0 += hv.x * w; a1 += hv.y * w; a2 += hv.z * w; a3 += hv.w * w;
  }
  float dv = dinv[n];
  float d2 = dv * dv;
  float4 hs = *(const float4*)(h + (size_t)n * DD + lane * 4);
  float4 bv = *(const float4*)(bias + lane * 4);
  a0 += hs.x * d2 + bv.x; a1 += hs.y * d2 + bv.y;
  a2 += hs.z * d2 + bv.z; a3 += hs.w * d2 + bv.w;
  *(float4*)(out + (size_t)n * DD + lane * 4) = make_float4(a0, a1, a2, a3);
}

// ---------------- per-graph GraphNorm (100 contiguous nodes, 128 dims) ----------------
__global__ __launch_bounds__(DD) void k_graphnorm(const float* __restrict__ xin,
                                                  const float* __restrict__ w,
                                                  const float* __restrict__ b,
                                                  const float* __restrict__ a,
                                                  float* __restrict__ out_f32,
                                                  _Float16* __restrict__ out_f16,
                                                  int do_relu) {
  int g = blockIdx.x, d = threadIdx.x;
  const float* base = xin + (size_t)g * NPG * DD + d;
  float s = 0.f;
  for (int i = 0; i < NPG; ++i) s += base[(size_t)i * DD];
  float mean = s * (1.0f / NPG);
  float ad = a[d], va = 0.f;
  for (int i = 0; i < NPG; ++i) { float t = base[(size_t)i * DD] - ad * mean; va += t * t; }
  float inv = rsqrtf(va * (1.0f / NPG) + GEPS);
  float wd = w[d], bd = b[d];
  for (int i = 0; i < NPG; ++i) {
    float t = base[(size_t)i * DD] - ad * mean;
    float o = wd * t * inv + bd;
    if (do_relu) o = fmaxf(o, 0.0f);
    size_t idx = (size_t)(g * NPG + i) * DD + d;
    if (out_f32) out_f32[idx] = o;
    if (out_f16) out_f16[idx] = (_Float16)o;
  }
}

// ---------------- W2 transpose+convert: Wt[n][k] = (f16) W[k][n] ----------------
__global__ void k_prep_w2(const float* __restrict__ W, _Float16* __restrict__ Wt) {
  int idx = blockIdx.x * blockDim.x + threadIdx.x;   // idx = k*128 + n
  if (idx >= DD * DD) return;
  int k = idx >> 7, n = idx & 127;
  Wt[(size_t)n * DD + k] = (_Float16)W[idx];
}

// ---------------- conv2 GEMM: C[M,128] = A[M,128] @ B[128,128]  via WMMA f16 ----------------
__global__ __launch_bounds__(256) void k_wmma_gemm(const _Float16* __restrict__ A,
                                                   const _Float16* __restrict__ Bt,
                                                   float* __restrict__ C) {
  int wave = threadIdx.x >> 5;        // 8 waves -> 8 column tiles (128/16)
  int lane = threadIdx.x & 31;
  int m0 = blockIdx.x * 16;
  int n0 = wave * 16;
  int mr = lane & 15;                 // A row within tile / B column within tile
  int hi = lane >> 4;
  const _Float16* Arow = A + (size_t)(m0 + mr) * DD;
  const _Float16* Brow = Bt + (size_t)(n0 + mr) * DD;
  __builtin_prefetch(A + (size_t)(m0 + 16 + mr) * DD, 0, 1);  // next M tile
  v8f acc = {};
  #pragma unroll
  for (int kc = 0; kc < 4; ++kc) {
    int ka = kc * 32 + hi * 8;                 // A 16-bit 16x32 layout
    v8h a_lo = *(const v8h*)(Arow + ka);
    v8h a_hi = *(const v8h*)(Arow + ka + 16);
    int kb = kc * 32 + hi * 8;                 // B 16-bit 32x16 layout
    v8h b_lo = *(const v8h*)(Brow + kb);
    v8h b_hi = *(const v8h*)(Brow + kb + 16);
    v16h av, bv;
    #pragma unroll
    for (int j = 0; j < 8; ++j) {
      av[j] = a_lo[j]; av[j + 8] = a_hi[j];
      bv[j] = b_lo[j]; bv[j + 8] = b_hi[j];
    }
    acc = __builtin_amdgcn_wmma_f32_16x16x32_f16(false, av, false, bv,
                                                 (short)0, acc, false, false);
  }
  int n = n0 + (lane & 15);
  int mb = m0 + hi * 8;
  #pragma unroll
  for (int r = 0; r < 8; ++r)
    C[(size_t)(mb + r) * DD + n] = acc[r];
}

// ---------------- Set2Set ----------------
__global__ void k_gates(const float* __restrict__ qstar, const float* __restrict__ h,
                        const float* __restrict__ Wih, const float* __restrict__ Whh,
                        const float* __restrict__ bih, const float* __restrict__ bhh,
                        float* __restrict__ gates) {
  int idx = blockIdx.x * blockDim.x + threadIdx.x;
  if (idx >= NB * 4 * DD) return;
  int g = idx >> 9, j = idx & 511;
  float s = bih[j] + bhh[j];
  const float* q = qstar + (size_t)g * 2 * DD;
  const float* wi = Wih + (size_t)j * 2 * DD;
  for (int k = 0; k < 2 * DD; ++k) s += q[k] * wi[k];
  const float* hh = h + (size_t)g * DD;
  const float* wh = Whh + (size_t)j * DD;
  for (int k = 0; k < DD; ++k) s += hh[k] * wh[k];
  gates[idx] = s;
}
__global__ void k_lstm(const float* __restrict__ gates, float* __restrict__ h,
                       float* __restrict__ c) {
  int idx = blockIdx.x * blockDim.x + threadIdx.x;
  if (idx >= NB * DD) return;
  int g = idx >> 7, d = idx & 127;
  const float* gb = gates + (size_t)g * 4 * DD;
  float ig = sigm(gb[d]), fg = sigm(gb[DD + d]);
  float gg = tanhf(gb[2 * DD + d]), og = sigm(gb[3 * DD + d]);
  float cv = fg * c[idx] + ig * gg;
  c[idx] = cv;
  h[idx] = og * tanhf(cv);
}
__global__ __launch_bounds__(256) void k_attn_e(const float* __restrict__ x,
                                                const float* __restrict__ h,
                                                const int* __restrict__ batch,
                                                float* __restrict__ e) {
  long long gid = (long long)blockIdx.x * blockDim.x + threadIdx.x;
  int n = (int)(gid >> 5);
  if (n >= NN) return;
  int lane = (int)(gid & 31);
  int g = batch[n];
  float4 xv = *(const float4*)(x + (size_t)n * DD + lane * 4);
  float4 hv = *(const float4*)(h + (size_t)g * DD + lane * 4);
  float p = xv.x * hv.x + xv.y * hv.y + xv.z * hv.z + xv.w * hv.w;
  p = wave_red_sum(p);
  if (lane == 0) e[n] = p;
}
__global__ __launch_bounds__(DD) void k_softmax_r(const float* __restrict__ e,
                                                  const float* __restrict__ x,
                                                  const float* __restrict__ h,
                                                  float* __restrict__ qstar) {
  __shared__ float se[NPG];
  __shared__ float sa[NPG];
  int g = blockIdx.x, d = threadIdx.x;
  if (d < NPG) se[d] = e[g * NPG + d];
  __syncthreads();
  float m = -3.0e38f;
  for (int i = 0; i < NPG; ++i) m = fmaxf(m, se[i]);
  float ssum = 0.f;
  for (int i = 0; i < NPG; ++i) ssum += expf(se[i] - m);
  if (d < NPG) sa[d] = expf(se[d] - m) / ssum;
  __syncthreads();
  const float* xb = x + (size_t)g * NPG * DD + d;
  float r = 0.f;
  for (int i = 0; i < NPG; ++i) r += sa[i] * xb[(size_t)i * DD];
  qstar[(size_t)g * 2 * DD + d] = h[(size_t)g * DD + d];
  qstar[(size_t)g * 2 * DD + DD + d] = r;
}

// ---------------- final MLP ----------------
__global__ void k_lin1(const float* __restrict__ qstar, const float* __restrict__ W,
                       const float* __restrict__ b, float* __restrict__ out) {
  int idx = blockIdx.x * blockDim.x + threadIdx.x;
  if (idx >= NB * DD) return;
  int g = idx >> 7, d = idx & 127;
  const float* q = qstar + (size_t)g * 2 * DD;
  float s = b[d];
  for (int k = 0; k < 2 * DD; ++k) s += q[k] * W[(size_t)k * DD + d];
  out[idx] = fmaxf(s, 0.0f);
}
__global__ void k_lin2(const float* __restrict__ t1, const float* __restrict__ W,
                       const float* __restrict__ b, float* __restrict__ out) {
  int g = blockIdx.x * blockDim.x + threadIdx.x;
  if (g >= NB) return;
  float s = b[0];
  const float* tb = t1 + (size_t)g * DD;
  for (int d = 0; d < DD; ++d) s += tb[d] * W[d];
  out[g] = s;
}

// =========================================================================
extern "C" void kernel_launch(void* const* d_in, const int* in_sizes, int n_in,
                              void* d_out, int out_size, void* d_ws, size_t ws_size,
                              hipStream_t stream) {
  (void)in_sizes; (void)n_in; (void)out_size; (void)ws_size;
  const float* x      = (const float*)d_in[0];
  const int*   ei     = (const int*)d_in[1];     // [2,E] flat
  const int*   batch  = (const int*)d_in[2];
  const float* gn0_w  = (const float*)d_in[3];
  const float* gn0_b  = (const float*)d_in[4];
  const float* gn0_a  = (const float*)d_in[5];
  const float* W1     = (const float*)d_in[6];
  const float* b1     = (const float*)d_in[7];
  const float* gn1_w  = (const float*)d_in[8];
  const float* gn1_b  = (const float*)d_in[9];
  const float* gn1_a  = (const float*)d_in[10];
  const float* W2     = (const float*)d_in[11];
  const float* b2     = (const float*)d_in[12];
  const float* gn2_w  = (const float*)d_in[13];
  const float* gn2_b  = (const float*)d_in[14];
  const float* gn2_a  = (const float*)d_in[15];
  const float* Wih    = (const float*)d_in[16];
  const float* Whh    = (const float*)d_in[17];
  const float* bih    = (const float*)d_in[18];
  const float* bhh    = (const float*)d_in[19];
  const float* l1W    = (const float*)d_in[20];
  const float* l1b    = (const float*)d_in[21];
  const float* l2W    = (const float*)d_in[22];
  const float* l2b    = (const float*)d_in[23];
  float* out = (float*)d_out;
  const int* esrc = ei;
  const int* edst = ei + NE;

  // ---- workspace carve-up (4-byte words) ----
  float* F = (float*)d_ws;
  size_t o = 0;
  float* xn0   = F + o; o += (size_t)NN * DIN;
  float* P0    = F + o; o += (size_t)NN * DD;    // GEMM result h1/h2
  float* P1    = F + o; o += (size_t)NN * DD;    // conv output
  float* P2    = F + o; o += (size_t)NN * DD;    // x after gn2
  float* dinv  = F + o; o += NN;
  float* stats = F + o; o += 8;
  float* hbuf  = F + o; o += (size_t)NB * DD;
  float* cbuf  = F + o; o += (size_t)NB * DD;
  float* qstar = F + o; o += (size_t)NB * 2 * DD;
  float* gates = F + o; o += (size_t)NB * 4 * DD;
  float* evec  = F + o; o += NN;
  float* t1    = F + o; o += (size_t)NB * DD;
  int*   degi   = (int*)(F + o); o += NN;
  int*   cursor = (int*)(F + o); o += NN;
  int*   rows   = (int*)(F + o); o += NN + 1;
  int*   bsum   = (int*)(F + o); o += NBLK;
  int*   boff   = (int*)(F + o); o += NBLK;
  float* cw     = F + o; o += NE;
  int*   csrc   = (int*)(F + o); o += NE;
  o = (o + 3) & ~(size_t)3;                      // 16B align for f16 vectors
  _Float16* x1h = (_Float16*)(F + o); o += (size_t)NN * DD / 2;
  _Float16* w2t = (_Float16*)(F + o); o += (size_t)DD * DD / 2;

  auto cdiv = [](long long a, long long b) { return (unsigned)((a + b - 1) / b); };

  // ---- GraphNorm0 (whole batch, D_IN=4) ----
  k_zero<<<1, 64, 0, stream>>>(stats, 8);
  k_gn0_sum<<<cdiv(NN, 256), 256, 0, stream>>>(x, stats);
  k_gn0_var<<<cdiv(NN, 256), 256, 0, stream>>>(x, gn0_a, stats);
  k_gn0_norm<<<cdiv(NN, 256), 256, 0, stream>>>(x, gn0_w, gn0_b, gn0_a, stats, xn0);

  // ---- CSR build (shared by both convs): deg -> scan -> dinv -> fill ----
  k_zero<<<cdiv(NN, 256), 256, 0, stream>>>((float*)degi, NN);
  k_zero<<<cdiv(NN, 256), 256, 0, stream>>>((float*)cursor, NN);
  k_deg_i<<<cdiv(NE, 256), 256, 0, stream>>>(edst, degi);
  k_scan1<<<NBLK, 256, 0, stream>>>(degi, rows, bsum);
  k_scan2<<<1, 512, 0, stream>>>(bsum, boff, NBLK);
  k_scan3<<<cdiv(NN, 256), 256, 0, stream>>>(rows, boff);
  k_dinv_i<<<cdiv(NN, 256), 256, 0, stream>>>(degi, dinv);
  k_fill<<<cdiv(NE, 256), 256, 0, stream>>>(esrc, edst, rows, cursor, dinv, csrc, cw);

  // ---- conv1: h1 = xn0 @ W1 ; CSR gather (+self+bias) ----
  k_conv1<<<NN, DD, 0, stream>>>(xn0, W1, P0);
  k_gather<<<cdiv((long long)NN * 32, 256), 256, 0, stream>>>(P0, rows, csrc, cw,
                                                              dinv, b1, P1);

  // ---- gn1 + relu -> x1h (f16 for WMMA) ----
  k_graphnorm<<<NB, DD, 0, stream>>>(P1, gn1_w, gn1_b, gn1_a, nullptr, x1h, 1);

  // ---- conv2: WMMA GEMM h2 = x1h @ W2 ; CSR gather (+self+bias) ----
  k_prep_w2<<<cdiv(DD * DD, 256), 256, 0, stream>>>(W2, w2t);
  k_wmma_gemm<<<NN / 16, 256, 0, stream>>>(x1h, w2t, P0);
  k_gather<<<cdiv((long long)NN * 32, 256), 256, 0, stream>>>(P0, rows, csrc, cw,
                                                              dinv, b2, P1);

  // ---- gn2 -> P2 ----
  k_graphnorm<<<NB, DD, 0, stream>>>(P1, gn2_w, gn2_b, gn2_a, P2, nullptr, 0);

  // ---- Set2Set (2 steps) ----
  k_zero<<<cdiv((long long)NB * DD, 256), 256, 0, stream>>>(hbuf, (long long)NB * DD);
  k_zero<<<cdiv((long long)NB * DD, 256), 256, 0, stream>>>(cbuf, (long long)NB * DD);
  k_zero<<<cdiv((long long)NB * 2 * DD, 256), 256, 0, stream>>>(qstar, (long long)NB * 2 * DD);
  for (int step = 0; step < 2; ++step) {
    k_gates<<<cdiv((long long)NB * 4 * DD, 256), 256, 0, stream>>>(qstar, hbuf, Wih, Whh,
                                                                   bih, bhh, gates);
    k_lstm<<<cdiv((long long)NB * DD, 256), 256, 0, stream>>>(gates, hbuf, cbuf);
    k_attn_e<<<cdiv((long long)NN * 32, 256), 256, 0, stream>>>(P2, hbuf, batch, evec);
    k_softmax_r<<<NB, DD, 0, stream>>>(evec, P2, hbuf, qstar);
  }

  // ---- final MLP ----
  k_lin1<<<cdiv((long long)NB * DD, 256), 256, 0, stream>>>(qstar, l1W, l1b, t1);
  k_lin2<<<cdiv(NB, 256), 256, 0, stream>>>(t1, l2W, l2b, out);
}
